// RPEMultiHeadAttention_24421184045598
// MI455X (gfx1250) — compile-verified, matrix operand order checked by reading, not run on hardware
//
#include <hip/hip_runtime.h>
#include <hip/hip_bf16.h>

typedef __attribute__((ext_vector_type(16))) __bf16 v16bf;
typedef __attribute__((ext_vector_type(8)))  __bf16 v8bf;
typedef __attribute__((ext_vector_type(8)))  float  v8f;
typedef __attribute__((ext_vector_type(4)))  float  v4f;

#define N_TOK    1008
#define D_MODEL  256
#define N_HEADS  8
#define DK       32
#define NT       14
#define NF       72
#define NDF      143          // 2*NF-1
#define NDT      27           // 2*NT-1
#define TBL_ROWS (NDT * NDF)  // 3861

static __device__ __forceinline__ v8f v8f_zero() {
  v8f z;
#pragma unroll
  for (int i = 0; i < 8; ++i) z[i] = 0.0f;
  return z;
}

static __device__ __forceinline__ v8f wmma_bf16(v16bf a, v16bf b, v8f c) {
  // D = A(16x32 bf16) * B(32x16 bf16) + C(16x16 f32)
  return __builtin_amdgcn_wmma_f32_16x16x32_bf16(
      /*neg_a=*/false, a, /*neg_b=*/false, b,
      /*c_mod=*/(short)0, c, /*reuse_a=*/false, /*reuse_b=*/false);
}

// CDNA5 wave32 16-bit fragment from a contiguous row: this lane supplies
// K = [kb..kb+7] and [kb+16..kb+23] of the 32-wide K tile. Both chunks are
// 16B-aligned contiguous runs -> two global_load_b128 / ds_load_b128.
static __device__ __forceinline__ v16bf frag_row_bf16(const __bf16* row, int kb) {
  v8bf lo = *(const v8bf*)(row + kb);
  v8bf hi = *(const v8bf*)(row + kb + 16);
  v16bf f;
#pragma unroll
  for (int i = 0; i < 8; ++i) { f[i] = lo[i]; f[8 + i] = hi[i]; }
  return f;
}
// Same, converting from f32 in-register (two 32B vector loads + pack-cvt).
static __device__ __forceinline__ v16bf frag_row_f32(const float* row, int kb) {
  v4f a0 = *(const v4f*)(row + kb);
  v4f a1 = *(const v4f*)(row + kb + 4);
  v4f b0 = *(const v4f*)(row + kb + 16);
  v4f b1 = *(const v4f*)(row + kb + 20);
  v16bf f;
#pragma unroll
  for (int i = 0; i < 4; ++i) {
    f[i]      = (__bf16)a0[i];
    f[4 + i]  = (__bf16)a1[i];
    f[8 + i]  = (__bf16)b0[i];
    f[12 + i] = (__bf16)b1[i];
  }
  return f;
}

// ---------------------------------------------------------------------------
// RPE bias table: tbl[p][head], p = it*NDF + jf over (2Nt-1)x(2Nf-1) offsets.
// ---------------------------------------------------------------------------
__global__ void rpe_table_kernel(const float* __restrict__ W_rpe,
                                 float* __restrict__ tbl) {
  int p = blockIdx.x * blockDim.x + threadIdx.x;
  if (p >= TBL_ROWS) return;
  int it = p / NDF, jf = p % NDF;
  float dt = (float)(it - (NT - 1)) / (float)(NT - 1);
  float df = (float)(jf - (NF - 1)) / (float)(NF - 1);
  float enc[32];
#pragma unroll
  for (int j = 0; j < 8; ++j) {
    float fr = __expf(-9.210340371976184f * (float)j / 8.0f); // 10000^(-j/8)
    float a0 = dt * fr, a1 = df * fr;
    enc[j]      = __sinf(a0);
    enc[8 + j]  = __cosf(a0);
    enc[16 + j] = __sinf(a1);
    enc[24 + j] = __cosf(a1);
  }
#pragma unroll
  for (int hh = 0; hh < N_HEADS; ++hh) {
    float s = 0.0f;
#pragma unroll
    for (int c = 0; c < 32; ++c) s += enc[c] * W_rpe[hh * 32 + c];
    tbl[p * N_HEADS + hh] = s;
  }
}

// ---------------------------------------------------------------------------
// QKV projection: Out[b,h,n,d] (bf16) = (X @ W^T), X f32 [8064,256], W f32
// [256,256] row-major (so B[k,n] = W[n,k] -> per-lane contiguous loads).
// Block = 128 threads (4 waves); wave w covers a 16x64 output strip.
// ---------------------------------------------------------------------------
__global__ void __launch_bounds__(128)
proj_qkv_wmma(const float* __restrict__ X, const float* __restrict__ W,
              __bf16* __restrict__ Out) {
  const int lane  = threadIdx.x & 31;
  const int wave  = threadIdx.x >> 5;     // 0..3
  const int m0    = blockIdx.x * 16;      // row tile (aligned: 1008 % 16 == 0)
  const int n0    = wave * 64;
  const int kb    = (lane & 16) ? 8 : 0;
  const int mr    = lane & 15;
  const int rbase = (lane & 16) ? 8 : 0;

  v8f acc[4];
#pragma unroll
  for (int t = 0; t < 4; ++t) acc[t] = v8f_zero();

  const float* xrow = X + (size_t)(m0 + mr) * D_MODEL;
  const float* wrow0 = W + (size_t)(n0 + mr) * D_MODEL;

  for (int k0 = 0; k0 < D_MODEL; k0 += 32) {
    if (k0 + 32 < D_MODEL) {              // gfx1250 global_prefetch_b8
      __builtin_prefetch(xrow + k0 + 32, 0, 3);
      __builtin_prefetch(wrow0 + k0 + 32, 0, 3);
    }
    v16bf a = frag_row_f32(xrow + k0, kb);
#pragma unroll
    for (int t = 0; t < 4; ++t) {
      v16bf b = frag_row_f32(wrow0 + (size_t)t * 16 * D_MODEL + k0, kb);
      acc[t] = wmma_bf16(a, b, acc[t]);
    }
  }

  const int bidx  = m0 / N_TOK;
  const int nrow0 = m0 % N_TOK;
#pragma unroll
  for (int t = 0; t < 4; ++t) {
    int j = n0 + t * 16 + mr;          // output feature 0..255
    int hh = j >> 5, dd = j & 31;      // head, dim-in-head
    size_t obase = ((size_t)bidx * N_HEADS + hh) * (size_t)N_TOK * DK + dd;
#pragma unroll
    for (int r = 0; r < 8; ++r) {
      int q = nrow0 + rbase + r;
      Out[obase + (size_t)q * DK] = (__bf16)acc[t][r];
    }
  }
}

// ---------------------------------------------------------------------------
// Flash attention with RPE bias. One wave per 16-query tile per (b,h).
// Per 32-key chunk: async-DMA the V tile into LDS (ASYNCcnt path), 2 WMMAs
// for scores, online softmax (shfl reductions), P via LDS -> A fragment,
// s_wait_asynccnt, 2 WMMAs for P@V.
// ---------------------------------------------------------------------------
__global__ void __launch_bounds__(32)
flash_attn_rpe_wmma(const __bf16* __restrict__ Q, const __bf16* __restrict__ K,
                    const __bf16* __restrict__ V, const float* __restrict__ tbl,
                    const int* __restrict__ t_q, const int* __restrict__ f_q,
                    const int* __restrict__ t_k, const int* __restrict__ f_k,
                    __bf16* __restrict__ O) {
  __shared__ __align__(16) __bf16 ldsP[16 * 32];   // P tile (C->A relayout)
  __shared__ __align__(16) __bf16 ldsV[32 * 32];   // staged V tile
  const int lane  = threadIdx.x & 31;
  const int qt    = blockIdx.x;       // 0..62
  const int bh    = blockIdx.y;       // 0..63 = b*8 + h
  const int b     = bh >> 3, h = bh & 7;
  const int q0    = qt * 16;
  const int kb    = (lane & 16) ? 8 : 0;
  const int coln  = lane & 15;
  const int rbase = (lane & 16) ? 8 : 0;
  const size_t base = (size_t)bh * N_TOK * DK;
  const float inv_scale = 0.1767766952966369f; // 1/sqrt(dk)

  // Per-lane LDS byte offset of this lane's V row. Per the ISA aperture rule
  // the low 32 bits of the generic pointer are
  // the LDS offset the async-DMA instruction wants in its VDST VGPR.
  unsigned ldsVOff = (unsigned)(unsigned long long)(uintptr_t)&ldsV[lane * 32];

  // Q A-fragment: loaded once, reused for all 64 score WMMAs of this tile.
  v16bf aq = frag_row_bf16(Q + base + (size_t)(q0 + coln) * DK, kb);

  v8f acc0 = v8f_zero(), acc1 = v8f_zero();
  float m_r[8], l_r[8];
  int tq_r[8], fq_r[8];
#pragma unroll
  for (int r = 0; r < 8; ++r) {
    m_r[r] = -1e30f; l_r[r] = 0.0f;
    int q = q0 + rbase + r;
    tq_r[r] = t_q[q]; fq_r[r] = f_q[q];
  }

  for (int k0 = 0; k0 < 1024; k0 += 32) {   // keys padded 1008 -> 1024
    // Prefetch next chunk's K rows (one row per lane covers the tile).
    if (k0 + 32 < N_TOK)
      __builtin_prefetch(K + base + (size_t)(k0 + 32 + lane) * DK, 0, 3);

    // Async-stage V tile rows k0..k0+31 (clamped) into LDS: no VGPR round
    // trip, tracked by ASYNCcnt, overlaps with the score WMMAs below.
    // INST_OFFSET is added to BOTH the global and LDS addresses (ISA 08 §4.4),
    // so one address pair + offset:0/16/32/48 moves the whole 64B row.
    {
      unsigned long long ga = (unsigned long long)(uintptr_t)
          (V + base + (size_t)min(k0 + lane, N_TOK - 1) * DK);
      asm volatile(
          "global_load_async_to_lds_b128 %0, %1, off\n\t"
          "global_load_async_to_lds_b128 %0, %1, off offset:16\n\t"
          "global_load_async_to_lds_b128 %0, %1, off offset:32\n\t"
          "global_load_async_to_lds_b128 %0, %1, off offset:48"
          :: "v"(ldsVOff), "v"(ga) : "memory");
    }

    int kc0 = k0 + coln, kc1 = k0 + 16 + coln;
    int kc0c = min(kc0, N_TOK - 1), kc1c = min(kc1, N_TOK - 1);

    // B = K^T: lane's column is one key token, contiguous over d (b128 loads).
    v16bf bk0 = frag_row_bf16(K + base + (size_t)kc0c * DK, kb);
    v16bf bk1 = frag_row_bf16(K + base + (size_t)kc1c * DK, kb);
    v8f s0 = wmma_bf16(aq, bk0, v8f_zero());
    v8f s1 = wmma_bf16(aq, bk1, v8f_zero());

    int tk0 = t_k[kc0c], fk0 = f_k[kc0c];
    int tk1 = t_k[kc1c], fk1 = f_k[kc1c];
#pragma unroll
    for (int r = 0; r < 8; ++r) {
      int i0 = min(max((tq_r[r] - tk0 + NT - 1) * NDF + (fq_r[r] - fk0 + NF - 1), 0),
                   TBL_ROWS - 1);
      int i1 = min(max((tq_r[r] - tk1 + NT - 1) * NDF + (fq_r[r] - fk1 + NF - 1), 0),
                   TBL_ROWS - 1);
      float v0 = s0[r] * inv_scale + tbl[i0 * N_HEADS + h];
      float v1 = s1[r] * inv_scale + tbl[i1 * N_HEADS + h];
      s0[r] = (kc0 < N_TOK) ? v0 : -1e30f;
      s1[r] = (kc1 < N_TOK) ? v1 : -1e30f;
    }

    // Online softmax update; rows live in 16-lane halves -> xor masks <= 8.
#pragma unroll
    for (int r = 0; r < 8; ++r) {
      float mx = fmaxf(s0[r], s1[r]);
#pragma unroll
      for (int off = 8; off >= 1; off >>= 1) mx = fmaxf(mx, __shfl_xor(mx, off, 32));
      float mnew = fmaxf(m_r[r], mx);
      float sc   = __expf(m_r[r] - mnew);
      m_r[r] = mnew;
      float p0 = __expf(s0[r] - mnew);
      float p1 = __expf(s1[r] - mnew);
      float ps = p0 + p1;
#pragma unroll
      for (int off = 8; off >= 1; off >>= 1) ps += __shfl_xor(ps, off, 32);
      l_r[r] = l_r[r] * sc + ps;
      acc0[r] = acc0[r] * sc;
      acc1[r] = acc1[r] * sc;
      int m = rbase + r;
      ldsP[m * 32 + coln]      = (__bf16)p0;  // C layout -> row-major 16x32
      ldsP[m * 32 + 16 + coln] = (__bf16)p1;
    }
    __syncthreads();   // single-wave WG: barrier is a NOP, forces DS waits
    asm volatile("s_wait_asynccnt 0x0" ::: "memory");  // V tile landed in LDS

    v16bf ap = frag_row_bf16(&ldsP[coln * 32], kb);   // P as A fragment

    // B = V tile from LDS (K-strided 16-bit reads stay on the DS path).
    v16bf bv0, bv1;
#pragma unroll
    for (int i = 0; i < 8; ++i) {
      int r0 = (kb + i) * 32;
      int r1 = (kb + 16 + i) * 32;
      bv0[i]     = ldsV[r0 + coln];
      bv0[8 + i] = ldsV[r1 + coln];
      bv1[i]     = ldsV[r0 + 16 + coln];
      bv1[8 + i] = ldsV[r1 + 16 + coln];
    }
    acc0 = wmma_bf16(ap, bv0, acc0);
    acc1 = wmma_bf16(ap, bv1, acc1);
    __syncthreads();
  }

#pragma unroll
  for (int r = 0; r < 8; ++r) {
    float inv_l = 1.0f / l_r[r];
    int q = q0 + rbase + r;
    size_t o = ((size_t)b * N_TOK + q) * D_MODEL + h * DK;
    O[o + coln]      = (__bf16)(acc0[r] * inv_l);
    O[o + 16 + coln] = (__bf16)(acc1[r] * inv_l);
  }
}

// ---------------------------------------------------------------------------
// Output projection: d_out (f32 [8064,256]) = Obf (bf16) @ W_o^T.
// ---------------------------------------------------------------------------
__global__ void __launch_bounds__(128)
out_proj_wmma(const __bf16* __restrict__ X, const float* __restrict__ W,
              float* __restrict__ Out) {
  const int lane  = threadIdx.x & 31;
  const int wave  = threadIdx.x >> 5;
  const int m0    = blockIdx.x * 16;
  const int n0    = wave * 64;
  const int kb    = (lane & 16) ? 8 : 0;
  const int mr    = lane & 15;
  const int rbase = (lane & 16) ? 8 : 0;

  v8f acc[4];
#pragma unroll
  for (int t = 0; t < 4; ++t) acc[t] = v8f_zero();

  const __bf16* xrow = X + (size_t)(m0 + mr) * D_MODEL;
  const float* wrow0 = W + (size_t)(n0 + mr) * D_MODEL;

  for (int k0 = 0; k0 < D_MODEL; k0 += 32) {
    if (k0 + 32 < D_MODEL) {
      __builtin_prefetch(xrow + k0 + 32, 0, 3);
      __builtin_prefetch(wrow0 + k0 + 32, 0, 3);
    }
    v16bf a = frag_row_bf16(xrow + k0, kb);
#pragma unroll
    for (int t = 0; t < 4; ++t) {
      v16bf bmat = frag_row_f32(wrow0 + (size_t)t * 16 * D_MODEL + k0, kb);
      acc[t] = wmma_bf16(a, bmat, acc[t]);
    }
  }
#pragma unroll
  for (int t = 0; t < 4; ++t) {
    int j = n0 + t * 16 + mr;
#pragma unroll
    for (int r = 0; r < 8; ++r)
      Out[(size_t)(m0 + rbase + r) * D_MODEL + j] = acc[t][r];
  }
}

extern "C" void kernel_launch(void* const* d_in, const int* in_sizes, int n_in,
                              void* d_out, int out_size, void* d_ws, size_t ws_size,
                              hipStream_t stream) {
  (void)in_sizes; (void)n_in; (void)out_size; (void)ws_size;
  const float* q_tokens = (const float*)d_in[0];
  const float* k_tokens = (const float*)d_in[1];
  const float* v_tokens = (const float*)d_in[2];
  const int*   t_q      = (const int*)d_in[3];
  const int*   f_q      = (const int*)d_in[4];
  const int*   t_k      = (const int*)d_in[5];
  const int*   f_k      = (const int*)d_in[6];
  // d_in[7]=N_t, d_in[8]=N_f (compile-time constants here: 14, 72)
  const float* W_q   = (const float*)d_in[9];
  const float* W_k   = (const float*)d_in[10];
  const float* W_v   = (const float*)d_in[11];
  const float* W_o   = (const float*)d_in[12];
  const float* W_rpe = (const float*)d_in[13];

  // Workspace: | tbl f32 (128KB) | Qbf | Kbf | Vbf | Obf |  (bf16, 4MB each)
  float*  tbl = (float*)d_ws;
  __bf16* Qbf = (__bf16*)((char*)d_ws + 131072);
  const size_t ntot = (size_t)8 * N_TOK * D_MODEL;   // 2,064,384 elements
  __bf16* Kbf = Qbf + ntot;
  __bf16* Vbf = Kbf + ntot;
  __bf16* Obf = Vbf + ntot;

  rpe_table_kernel<<<(TBL_ROWS + 127) / 128, 128, 0, stream>>>(W_rpe, tbl);
  proj_qkv_wmma<<<504, 128, 0, stream>>>(q_tokens, W_q, Qbf);
  proj_qkv_wmma<<<504, 128, 0, stream>>>(k_tokens, W_k, Kbf);
  proj_qkv_wmma<<<504, 128, 0, stream>>>(v_tokens, W_v, Vbf);
  flash_attn_rpe_wmma<<<dim3(63, 64), 32, 0, stream>>>(Qbf, Kbf, Vbf, tbl,
                                                       t_q, f_q, t_k, f_k, Obf);
  out_proj_wmma<<<504, 128, 0, stream>>>(Obf, W_o, (float*)d_out);
}
